// GuideAttentionModule_56375740727765
// MI455X (gfx1250) — compile-verified, matrix operand order checked by reading, not run on hardware
//
#include <hip/hip_runtime.h>
#include <hip/hip_bf16.h>

typedef __attribute__((ext_vector_type(16))) __bf16 v16bf;
typedef __attribute__((ext_vector_type(8)))  __bf16 v8bf;
typedef __attribute__((ext_vector_type(8)))  float  v8f;
typedef __attribute__((ext_vector_type(4)))  int    v4i;

#define D_MODEL 1024
#define NUM_HEADS 16
#define HEAD_DIM 64
#define BATCH 4
#define SEQ 1024
#define MTOT (BATCH * SEQ)   // 4096

// ---- optional CDNA5 async global->LDS path --------------------------------
#if defined(__has_builtin)
#  if __has_builtin(__builtin_amdgcn_global_load_async_to_lds_b128)
#    define HAVE_ASYNC_LDS 1
#  endif
#  if __has_builtin(__builtin_amdgcn_s_wait_asynccnt)
#    define HAVE_WAIT_ASYNC_BUILTIN 1
#  endif
#endif
#ifndef HAVE_ASYNC_LDS
#  define HAVE_ASYNC_LDS 0
#endif
#ifndef HAVE_WAIT_ASYNC_BUILTIN
#  define HAVE_WAIT_ASYNC_BUILTIN 0
#endif

// copy 16 bytes (8 bf16) global -> LDS
__device__ __forceinline__ void stage16(const __bf16* gsrc, __bf16* lds_dst) {
#if HAVE_ASYNC_LDS
    __builtin_amdgcn_global_load_async_to_lds_b128(
        (__attribute__((address_space(1))) v4i*)(gsrc),
        (__attribute__((address_space(3))) v4i*)(lds_dst), 0, 0);
#else
    *reinterpret_cast<v8bf*>(lds_dst) = *reinterpret_cast<const v8bf*>(gsrc);
#endif
}

__device__ __forceinline__ void wait_async() {
#if HAVE_ASYNC_LDS
#  if HAVE_WAIT_ASYNC_BUILTIN
    __builtin_amdgcn_s_wait_asynccnt(0);
#  else
    asm volatile("s_wait_asynccnt 0" ::: "memory");
#  endif
#endif
}

// ---------------------------------------------------------------------------
// bf16 WMMA fragment helpers (CDNA5 ISA 7.12.2 layouts)
// A/B fragment: lane l holds row/col (l&15);
// elements 0..7  = K + 8*(l>>4) + 0..7 ; elements 8..15 = K + 16 + 8*(l>>4)
// ---------------------------------------------------------------------------
__device__ __forceinline__ v16bf load_frag(const __bf16* rowBase, int kk, int half) {
    const v8bf lo = *reinterpret_cast<const v8bf*>(rowBase + kk + 8 * half);
    const v8bf hi = *reinterpret_cast<const v8bf*>(rowBase + kk + 16 + 8 * half);
    v16bf r;
#pragma unroll
    for (int i = 0; i < 8; ++i) { r[i] = lo[i]; r[8 + i] = hi[i]; }
    return r;
}

__device__ __forceinline__ v8f wmma_bf16(v16bf a, v16bf b, v8f c) {
    return __builtin_amdgcn_wmma_f32_16x16x32_bf16(
        false, a, false, b, (short)0, c, false, false);
}

// ---------------------------------------------------------------------------
// Kernel 0: fp32 -> bf16 cast
// ---------------------------------------------------------------------------
__global__ void cast_bf16_kernel(const float* __restrict__ src,
                                 __bf16* __restrict__ dst, int n) {
    int i = blockIdx.x * blockDim.x + threadIdx.x;
    if (i < n) dst[i] = (__bf16)src[i];
}

// ---------------------------------------------------------------------------
// Kernel 1: token projections (fp32, tiny)
// ---------------------------------------------------------------------------
__global__ void token_proj_kernel(const float* __restrict__ tokens,
                                  const float* __restrict__ Wq,
                                  const float* __restrict__ bq,
                                  const float* __restrict__ Wk,
                                  const float* __restrict__ bk,
                                  float* __restrict__ tQf,
                                  float* __restrict__ tKf) {
    int idx = blockIdx.x * blockDim.x + threadIdx.x;
    if (idx >= 2 * BATCH * D_MODEL) return;
    int which = idx >> 12;
    int b = (idx >> 10) & (BATCH - 1);
    int n = idx & (D_MODEL - 1);
    const float* W = which ? Wk : Wq;
    const float* bias = which ? bk : bq;
    const float* t = tokens + b * D_MODEL;
    const float* w = W + n * D_MODEL;
    float acc = bias[n];
#pragma unroll 4
    for (int k = 0; k < D_MODEL; ++k) acc += t[k] * w[k];
    (which ? tKf : tQf)[b * D_MODEL + n] = acc;
}

// ---------------------------------------------------------------------------
// Kernel 2: QKV projection GEMM.
// 256 threads = 8 waves; block tile 128M x 64N (wave w: rows 16w..16w+15).
// B (64 rows x 32 K) staged into double-buffered LDS, shared by all 8 waves.
// A fragments register-double-buffered. z selects Q/K/V epilogue.
// ---------------------------------------------------------------------------
__global__ __launch_bounds__(256) void proj_kernel(
    const __bf16* __restrict__ xb,
    const __bf16* __restrict__ Wqb, const __bf16* __restrict__ Wkb,
    const __bf16* __restrict__ Wvb,
    const float* __restrict__ bq, const float* __restrict__ bk,
    const float* __restrict__ bv,
    const float* __restrict__ tQf, const float* __restrict__ tKf,
    __bf16* __restrict__ Qb, __bf16* __restrict__ Kb,
    __bf16* __restrict__ Vt) {
    __shared__ __bf16 Blds[2][64 * 32];

    const int tid = threadIdx.x;
    const int lane = tid & 31;
    const int waveId = tid >> 5;
    const int half = lane >> 4;
    const int l15  = lane & 15;
    const int m0w = blockIdx.x * 128 + waveId * 16;
    const int n0 = blockIdx.y * 64;
    const int z  = blockIdx.z;

    const __bf16* W = (z == 0) ? Wqb : ((z == 1) ? Wkb : Wvb);

    // staging assignment: 256 segs of 16B (64 rows x 4 segs)
    const int srow = tid >> 2;
    const int soff = (tid & 3) * 8;
    const __bf16* Wsrc = W + (size_t)(n0 + srow) * D_MODEL + soff;
    __bf16* Bdst0 = &Blds[0][srow * 32 + soff];
    __bf16* Bdst1 = &Blds[1][srow * 32 + soff];

    stage16(Wsrc, Bdst0);
    wait_async();
    __syncthreads();

    const __bf16* Arow = xb + (size_t)(m0w + l15) * D_MODEL;
    v16bf a_cur = load_frag(Arow, 0, half);

    v8f acc[4];
#pragma unroll
    for (int t = 0; t < 4; ++t)
#pragma unroll
        for (int v = 0; v < 8; ++v) acc[t][v] = 0.f;

    int p = 0;
    for (int kk = 0; kk < D_MODEL; kk += 32) {
        const bool more = (kk + 32) < D_MODEL;
        if (more) stage16(Wsrc + kk + 32, p ? Bdst0 : Bdst1);   // fill p^1
        v16bf a_use = a_cur;
        if (more) a_cur = load_frag(Arow, kk + 32, half);

#pragma unroll
        for (int t = 0; t < 4; ++t) {
            v16bf bf = load_frag(&Blds[p][(16 * t + l15) * 32], 0, half);
            acc[t] = wmma_bf16(a_use, bf, acc[t]);
        }
        wait_async();
        __syncthreads();
        p ^= 1;
    }

    const int b = m0w >> 10;
    const int sbase = m0w & (SEQ - 1);
    const float* bias = (z == 0) ? bq : ((z == 1) ? bk : bv);

#pragma unroll
    for (int t = 0; t < 4; ++t) {
        int col = n0 + 16 * t + l15;
        int h = col >> 6;
        int d = col & 63;
        float badd = bias[col];
        float tadd = (z == 0) ? tQf[b * D_MODEL + col]
                   : (z == 1) ? tKf[b * D_MODEL + col] : 0.f;
        float scl = (z == 0) ? 0.125f : 1.0f;   // fold 1/sqrt(hd) into Q
#pragma unroll
        for (int v = 0; v < 8; ++v) {
            int srowg = sbase + v + 8 * half;
            float val = (acc[t][v] + badd + tadd) * scl;
            if (z == 2) {
                Vt[((size_t)(b * NUM_HEADS + h) * HEAD_DIM + d) * SEQ + srowg] =
                    (__bf16)val;
            } else {
                __bf16* dst = (z == 0) ? Qb : Kb;
                dst[((size_t)(b * NUM_HEADS + h) * SEQ + srowg) * HEAD_DIM + d] =
                    (__bf16)val;
            }
        }
    }
}

// ---------------------------------------------------------------------------
// Kernel 3: flash attention. 128 threads = 4 waves; block = 64 queries of one
// (b,h). K/V 32-key chunks staged in double-buffered LDS shared by the waves.
// ---------------------------------------------------------------------------
__global__ __launch_bounds__(128) void attn_kernel(
    const __bf16* __restrict__ Qb, const __bf16* __restrict__ Kb,
    const __bf16* __restrict__ Vt, float* __restrict__ out) {
    __shared__ __bf16 Kst[2][32 * 64];   // [key-local][d]
    __shared__ __bf16 Vst[2][64 * 32];   // [d][key-local]
    __shared__ __bf16 Plds[4 * 16 * 32]; // per-wave P bounce

    const int tid = threadIdx.x;
    const int lane = tid & 31;
    const int waveId = tid >> 5;
    const int half = lane >> 4;
    const int l15  = lane & 15;
    const int s0 = blockIdx.x * 64 + waveId * 16;
    const int h  = blockIdx.y;
    const int b  = blockIdx.z;
    const int bh = b * NUM_HEADS + h;

    // staging: K has 256 16B segs ([32 keys][8 segs of d]); V has 256
    // ([64 d][4 segs of keys]); each thread does 2 of each.
    const int kr0 = tid >> 3,         ko0 = (tid & 7) * 8;
    const int kr1 = (tid + 128) >> 3, ko1 = ((tid + 128) & 7) * 8;
    const int vr0 = tid >> 2,         vo0 = (tid & 3) * 8;
    const int vr1 = (tid + 128) >> 2, vo1 = ((tid + 128) & 3) * 8;

    const __bf16* Kbase = Kb + (size_t)bh * SEQ * HEAD_DIM;
    const __bf16* Vbase = Vt + (size_t)bh * HEAD_DIM * SEQ;

    // prologue: stage chunk j=0 into buffer 0
    stage16(Kbase + (size_t)(0 + kr0) * HEAD_DIM + ko0, &Kst[0][kr0 * 64 + ko0]);
    stage16(Kbase + (size_t)(0 + kr1) * HEAD_DIM + ko1, &Kst[0][kr1 * 64 + ko1]);
    stage16(Vbase + (size_t)vr0 * SEQ + 0 + vo0,        &Vst[0][vr0 * 32 + vo0]);
    stage16(Vbase + (size_t)vr1 * SEQ + 0 + vo1,        &Vst[0][vr1 * 32 + vo1]);
    wait_async();
    __syncthreads();

    const __bf16* Qrow = Qb + ((size_t)bh * SEQ + s0 + l15) * HEAD_DIM;
    const v16bf qa0 = load_frag(Qrow, 0,  half);
    const v16bf qa1 = load_frag(Qrow, 32, half);

    float mrow[8], lrow[8];
    v8f o[4];
#pragma unroll
    for (int v = 0; v < 8; ++v) { mrow[v] = -3.0e38f; lrow[v] = 0.f; }
#pragma unroll
    for (int t = 0; t < 4; ++t)
#pragma unroll
        for (int v = 0; v < 8; ++v) o[t][v] = 0.f;

    __bf16* Pw = &Plds[waveId * 16 * 32];

    int p = 0;
    for (int j = 0; j < SEQ; j += 32) {
        const bool more = (j + 32) < SEQ;
        if (more) {
            const int jn = j + 32;
            __bf16* kd = p ? &Kst[0][0] : &Kst[1][0];
            __bf16* vd = p ? &Vst[0][0] : &Vst[1][0];
            stage16(Kbase + (size_t)(jn + kr0) * HEAD_DIM + ko0, kd + kr0 * 64 + ko0);
            stage16(Kbase + (size_t)(jn + kr1) * HEAD_DIM + ko1, kd + kr1 * 64 + ko1);
            stage16(Vbase + (size_t)vr0 * SEQ + jn + vo0,        vd + vr0 * 32 + vo0);
            stage16(Vbase + (size_t)vr1 * SEQ + jn + vo1,        vd + vr1 * 32 + vo1);
        }

        // ---- scores for keys [j, j+32): two 16x16 tiles from LDS ----------
        v8f sa, sb;
#pragma unroll
        for (int v = 0; v < 8; ++v) { sa[v] = 0.f; sb[v] = 0.f; }
        sa = wmma_bf16(qa0, load_frag(&Kst[p][l15 * 64 + 0],  0, half), sa);
        sa = wmma_bf16(qa1, load_frag(&Kst[p][l15 * 64 + 32], 0, half), sa);
        sb = wmma_bf16(qa0, load_frag(&Kst[p][(16 + l15) * 64 + 0],  0, half), sb);
        sb = wmma_bf16(qa1, load_frag(&Kst[p][(16 + l15) * 64 + 32], 0, half), sb);

        // ---- online softmax ------------------------------------------------
        float mnew[8], alpha[8];
#pragma unroll
        for (int v = 0; v < 8; ++v) {
            float t = fmaxf(sa[v], sb[v]);
            t = fmaxf(t, __shfl_xor(t, 1, 32));
            t = fmaxf(t, __shfl_xor(t, 2, 32));
            t = fmaxf(t, __shfl_xor(t, 4, 32));
            t = fmaxf(t, __shfl_xor(t, 8, 32));
            mnew[v]  = fmaxf(mrow[v], t);
            alpha[v] = __expf(mrow[v] - mnew[v]);
            mrow[v]  = mnew[v];
        }
        float pa[8], pb[8];
#pragma unroll
        for (int v = 0; v < 8; ++v) {
            pa[v] = __expf(sa[v] - mnew[v]);
            pb[v] = __expf(sb[v] - mnew[v]);
            float rs = pa[v] + pb[v];
            rs += __shfl_xor(rs, 1, 32);
            rs += __shfl_xor(rs, 2, 32);
            rs += __shfl_xor(rs, 4, 32);
            rs += __shfl_xor(rs, 8, 32);
            lrow[v] = lrow[v] * alpha[v] + rs;
        }
#pragma unroll
        for (int t = 0; t < 4; ++t)
#pragma unroll
            for (int v = 0; v < 8; ++v) o[t][v] *= alpha[v];

        // ---- P: C-layout -> A-layout via per-wave LDS region --------------
#pragma unroll
        for (int v = 0; v < 8; ++v) {
            int r = v + 8 * half;
            Pw[r * 32 + l15]      = (__bf16)pa[v];
            Pw[r * 32 + 16 + l15] = (__bf16)pb[v];
        }
        asm volatile("s_wait_dscnt 0" ::: "memory");
        v16bf pA = load_frag(&Pw[l15 * 32], 0, half);

        // ---- O += P @ V ----------------------------------------------------
#pragma unroll
        for (int t = 0; t < 4; ++t) {
            o[t] = wmma_bf16(pA, load_frag(&Vst[p][(16 * t + l15) * 32], 0, half),
                             o[t]);
        }

        wait_async();
        __syncthreads();
        p ^= 1;
    }

    // ---- normalize and store [B,S,D] fp32 ----------------------------------
#pragma unroll
    for (int t = 0; t < 4; ++t) {
        int col = h * HEAD_DIM + 16 * t + l15;
#pragma unroll
        for (int v = 0; v < 8; ++v) {
            int srow = s0 + v + 8 * half;
            out[((size_t)b * SEQ + srow) * D_MODEL + col] = o[t][v] / lrow[v];
        }
    }
}

// ---------------------------------------------------------------------------
extern "C" void kernel_launch(void* const* d_in, const int* in_sizes, int n_in,
                              void* d_out, int out_size, void* d_ws, size_t ws_size,
                              hipStream_t stream) {
    const float* x      = (const float*)d_in[0];
    const float* tokens = (const float*)d_in[1];
    const float* Wq     = (const float*)d_in[2];
    const float* bq     = (const float*)d_in[3];
    const float* Wk     = (const float*)d_in[4];
    const float* bk     = (const float*)d_in[5];
    const float* Wv     = (const float*)d_in[6];
    const float* bv     = (const float*)d_in[7];
    float* out = (float*)d_out;

    char* p = (char*)d_ws;
    __bf16* xb  = (__bf16*)p; p += (size_t)MTOT * D_MODEL * 2;
    __bf16* Wqb = (__bf16*)p; p += (size_t)D_MODEL * D_MODEL * 2;
    __bf16* Wkb = (__bf16*)p; p += (size_t)D_MODEL * D_MODEL * 2;
    __bf16* Wvb = (__bf16*)p; p += (size_t)D_MODEL * D_MODEL * 2;
    __bf16* Qb  = (__bf16*)p; p += (size_t)MTOT * D_MODEL * 2;
    __bf16* Kb  = (__bf16*)p; p += (size_t)MTOT * D_MODEL * 2;
    __bf16* Vt  = (__bf16*)p; p += (size_t)MTOT * D_MODEL * 2;
    float*  tQf = (float*)p;  p += (size_t)BATCH * D_MODEL * 4;
    float*  tKf = (float*)p;  p += (size_t)BATCH * D_MODEL * 4;

    {
        int n = MTOT * D_MODEL;
        cast_bf16_kernel<<<(n + 255) / 256, 256, 0, stream>>>(x, xb, n);
        n = D_MODEL * D_MODEL;
        cast_bf16_kernel<<<(n + 255) / 256, 256, 0, stream>>>(Wq, Wqb, n);
        cast_bf16_kernel<<<(n + 255) / 256, 256, 0, stream>>>(Wk, Wkb, n);
        cast_bf16_kernel<<<(n + 255) / 256, 256, 0, stream>>>(Wv, Wvb, n);
    }
    token_proj_kernel<<<(2 * BATCH * D_MODEL + 255) / 256, 256, 0, stream>>>(
        tokens, Wq, bq, Wk, bk, tQf, tKf);
    {
        dim3 grid(MTOT / 128, D_MODEL / 64, 3);
        proj_kernel<<<grid, 256, 0, stream>>>(xb, Wqb, Wkb, Wvb, bq, bk, bv,
                                              tQf, tKf, Qb, Kb, Vt);
    }
    {
        dim3 grid(SEQ / 64, NUM_HEADS, BATCH);
        attn_kernel<<<grid, 128, 0, stream>>>(Qb, Kb, Vt, out);
    }
}
